// Model_47502338294403
// MI455X (gfx1250) — matrix operand order repaired and checked by reading, not compiled, not run on hardware
//
#include <hip/hip_runtime.h>
#include <hip/hip_bf16.h>
#include <math.h>

// ---------------------------------------------------------------------------
// P-NET pathway MLP on MI455X (gfx1250, wave32).
// f16 WMMA GEMMs with f32 accumulation, LDS-staged double-buffered tiles
// fed by GLOBAL_LOAD_ASYNC_TO_LDS_B128 (ASYNCcnt) when available.
// ---------------------------------------------------------------------------

typedef __attribute__((ext_vector_type(16))) _Float16 v16h;
typedef __attribute__((ext_vector_type(8)))  float    v8f;
typedef int v4i __attribute__((vector_size(16)));   // matches builtin param type

#define GENES   9229
#define KMOL    3
#define IN_SIZE (GENES * KMOL)
#define BATCH   1024

#define TM 128          // block tile M
#define TN 128          // block tile N
#define LDSROW 40       // halves per LDS row: 32 K + 8 pad (80B, multiple of 16B)

#if defined(__has_builtin)
#if __has_builtin(__builtin_amdgcn_global_load_async_to_lds_b128)
#define USE_ASYNC_LDS 1
#endif
#endif

union FragU { v16h h; uint32_t u[8]; uint4 q[2]; };

// 16-byte global -> LDS copy (async when supported; sync fallback).
__device__ __forceinline__ void copy16(const _Float16* g, _Float16* l) {
#ifdef USE_ASYNC_LDS
  __builtin_amdgcn_global_load_async_to_lds_b128(
      (__attribute__((address_space(1))) v4i*)g,
      (__attribute__((address_space(3))) v4i*)l, 0, 0);
#else
  *(uint4*)l = *(const uint4*)g;
#endif
}

__device__ __forceinline__ void wait_async_le4() {
#ifdef USE_ASYNC_LDS
#if __has_builtin(__builtin_amdgcn_s_wait_asynccnt)
  __builtin_amdgcn_s_wait_asynccnt(4);
#else
  asm volatile("s_wait_asynccnt 0x4" ::: "memory");
#endif
#endif
}

__device__ __forceinline__ void wait_async_le0() {
#ifdef USE_ASYNC_LDS
#if __has_builtin(__builtin_amdgcn_s_wait_asynccnt)
  __builtin_amdgcn_s_wait_asynccnt(0);
#else
  asm volatile("s_wait_asynccnt 0x0" ::: "memory");
#endif
#endif
}

// ---------------------------------------------------------------------------
// Diagonal layer: h[b,g] = tanh(sum_k x[b,3g+k]*w0[3g+k] + b0[g]) -> f16,
// rows zero-padded to Kpad.
// ---------------------------------------------------------------------------
__global__ __launch_bounds__(256) void diag_kernel(
    const float* __restrict__ x, const float* __restrict__ w0,
    const float* __restrict__ b0, _Float16* __restrict__ A1, int Kpad) {
  size_t idx = (size_t)blockIdx.x * blockDim.x + threadIdx.x;
  size_t total = (size_t)BATCH * Kpad;
  if (idx >= total) return;
  int b = (int)(idx / Kpad);
  int g = (int)(idx % Kpad);
  _Float16 out = (_Float16)0.0f;
  if (g < GENES) {
    const float* xr = x + (size_t)b * IN_SIZE + 3 * g;
    const float* wr = w0 + 3 * g;
    float s = xr[0] * wr[0] + xr[1] * wr[1] + xr[2] * wr[2] + b0[g];
    out = (_Float16)tanhf(s);
  }
  A1[idx] = out;
}

// ---------------------------------------------------------------------------
// Weight prep: WT[n, k] = f16( W[k, n] * mask[k, n] ), zero-padded to
// [Npad, Kpad].  Masks are jax bool (1 byte).
// ---------------------------------------------------------------------------
__global__ __launch_bounds__(256) void prep_weight_kernel(
    const float* __restrict__ W, const unsigned char* __restrict__ m,
    _Float16* __restrict__ WT, int fi, int fo, int Kpad, int Npad) {
  size_t idx = (size_t)blockIdx.x * blockDim.x + threadIdx.x;
  size_t total = (size_t)Npad * Kpad;
  if (idx >= total) return;
  int n = (int)(idx / Kpad);
  int k = (int)(idx % Kpad);
  float v = 0.0f;
  if (n < fo && k < fi) {
    size_t src = (size_t)k * fo + n;
    v = m[src] ? W[src] : 0.0f;
  }
  WT[idx] = (_Float16)v;
}

// ---------------------------------------------------------------------------
// WMMA GEMM + bias + tanh with LDS double-buffered staging.
// Block = 256 threads = 8 waves (4 in M x 2 in N); block tile 128x128,
// per-wave tile 32x64 -> 2 A-frags x 4 B-frags -> 8 WMMAs per 32-K step.
// H[m,n] = tanh(sum_k A[m,k]*WT[n,k] + bias[n]); n >= fo stored as 0.
// ---------------------------------------------------------------------------
__global__ __launch_bounds__(256) void gemm_tanh_kernel(
    const _Float16* __restrict__ A, const _Float16* __restrict__ WT,
    const float* __restrict__ bias, _Float16* __restrict__ H,
    int Kpad, int Npad, int fo) {
  __shared__ __align__(16) _Float16 sA[2][TM * LDSROW];
  __shared__ __align__(16) _Float16 sB[2][TN * LDSROW];

  const int t    = threadIdx.x;
  const int lane = t & 31;
  const int wave = t >> 5;
  const int half = lane >> 4;
  const int r    = lane & 15;

  const int Mblk = blockIdx.y * TM;
  const int Nblk = blockIdx.x * TN;

  const int M0w = (wave & 3) * 32;   // wave offset in M within block tile
  const int N0w = (wave >> 2) * 64;  // wave offset in N within block tile

  // Copy assignment: each thread moves 32B of A and 32B of B per stage.
  const int crow = t >> 1;           // 0..127
  const int cch  = t & 1;            // which 32B half of the 64B K-row
  const _Float16* gA = A  + (size_t)(Mblk + crow) * Kpad + cch * 16;
  const _Float16* gB = WT + (size_t)(Nblk + crow) * Kpad + cch * 16;
  const int lOff = crow * LDSROW + cch * 16;

  auto issue = [&](int s) {
    const int k0 = s << 5;           // halves
    const int buf = s & 1;
    copy16(gA + k0,     &sA[buf][lOff]);
    copy16(gA + k0 + 8, &sA[buf][lOff + 8]);
    copy16(gB + k0,     &sB[buf][lOff]);
    copy16(gB + k0 + 8, &sB[buf][lOff + 8]);
  };

  v8f acc[2][4] = {};
  const int S = Kpad >> 5;

  issue(0);
  for (int s = 0; s < S; ++s) {
    if (s + 1 < S) { issue(s + 1); wait_async_le4(); }
    else          { wait_async_le0(); }
    __syncthreads();   // stage s visible to all waves

    const int buf = s & 1;
    // A fragments: lanes 0-15 hold K0-7 + K16-23, lanes 16-31 K8-15 + K24-31.
    v16h af[2];
#pragma unroll
    for (int im = 0; im < 2; ++im) {
      const _Float16* base = &sA[buf][(M0w + im * 16 + r) * LDSROW + 8 * half];
      FragU f;
      f.q[0] = *(const uint4*)(base);
      f.q[1] = *(const uint4*)(base + 16);
      af[im] = f.h;
    }
    // B fragments: lanes 0-15 hold K0-7 + K16-23, lanes 16-31 K8-15 + K24-31 (col = lane&15).
    v16h bf[4];
#pragma unroll
    for (int jn = 0; jn < 4; ++jn) {
      const _Float16* base = &sB[buf][(N0w + jn * 16 + r) * LDSROW + 8 * half];
      FragU f;
      f.q[0] = *(const uint4*)(base);
      f.q[1] = *(const uint4*)(base + 16);
      bf[jn] = f.h;
    }
#pragma unroll
    for (int im = 0; im < 2; ++im)
#pragma unroll
      for (int jn = 0; jn < 4; ++jn)
        acc[im][jn] = __builtin_amdgcn_wmma_f32_16x16x32_f16(
            false, af[im], false, bf[jn], (short)0, acc[im][jn], false, false);

    __syncthreads();   // all reads done before stage s+2 overwrites this buf
  }

  // Store per C/D layout: VGPR j -> row + j + 8*(lane>>4), col = lane&15.
  const int c = r;
#pragma unroll
  for (int im = 0; im < 2; ++im) {
#pragma unroll
    for (int jn = 0; jn < 4; ++jn) {
      const int n = Nblk + N0w + jn * 16 + c;
      const bool valid = (n < fo);
      const float bn = valid ? bias[n] : 0.0f;
#pragma unroll
      for (int j = 0; j < 8; ++j) {
        const int mrow = Mblk + M0w + im * 16 + j + 8 * half;
        float val = acc[im][jn][j] + bn;
        H[(size_t)mrow * Npad + n] = valid ? (_Float16)tanhf(val) : (_Float16)0.0f;
      }
    }
  }
}

// ---------------------------------------------------------------------------
// Sigmoid head: out[row*6+col] = sigmoid(dot(H[row,0:d], lw) + lb).
// One wave per batch row; shuffle reduction (wave32).
// ---------------------------------------------------------------------------
__global__ __launch_bounds__(256) void head_kernel(
    const _Float16* __restrict__ H, int Kpad, int d,
    const float* __restrict__ lw, const float* __restrict__ lb,
    float* __restrict__ out, int col) {
  const int lane = threadIdx.x & 31;
  const int wave = threadIdx.x >> 5;
  const int row = blockIdx.x * 8 + wave;
  if (row >= BATCH) return;
  const _Float16* hr = H + (size_t)row * Kpad;
  float acc = 0.0f;
  for (int k = lane; k < d; k += 32) acc += (float)hr[k] * lw[k];
#pragma unroll
  for (int off = 16; off > 0; off >>= 1) acc += __shfl_down(acc, off, 32);
  if (lane == 0) out[row * 6 + col] = 1.0f / (1.0f + expf(-(acc + lb[0])));
}

// ---------------------------------------------------------------------------
// Host-side orchestration
// ---------------------------------------------------------------------------
extern "C" void kernel_launch(void* const* d_in, const int* in_sizes, int n_in,
                              void* d_out, int out_size, void* d_ws, size_t ws_size,
                              hipStream_t stream) {
  (void)in_sizes; (void)n_in; (void)out_size; (void)ws_size;

  const float* x  = (const float*)d_in[0];
  const float* w0 = (const float*)d_in[1];
  const float* b0 = (const float*)d_in[2];
  const float*         W[5];
  const float*         bb[5];
  const unsigned char* mk[5];
  for (int i = 0; i < 5; ++i) {
    W[i]  = (const float*)d_in[3 + 3 * i];
    bb[i] = (const float*)d_in[4 + 3 * i];
    mk[i] = (const unsigned char*)d_in[5 + 3 * i];   // jax bool = 1 byte
  }
  const float* lw[6];
  const float* lb[6];
  for (int i = 0; i < 6; ++i) {
    lw[i] = (const float*)d_in[18 + 2 * i];
    lb[i] = (const float*)d_in[19 + 2 * i];
  }
  float* out = (float*)d_out;

  // Geometry: K padded to %32, N padded to %128 (no partial block tiles).
  // Note KPAD[i+1] == NPAD[i] so the activation chain strides line up.
  static const int FI[5]   = {9229, 1387, 1066, 447, 147};
  static const int FO[5]   = {1387, 1066,  447, 147,  26};
  static const int KPAD[5] = {9248, 1408, 1152, 512, 256};
  static const int NPAD[5] = {1408, 1152,  512, 256, 128};
  static const int DIMS[6] = {9229, 1387, 1066, 447, 147, 26};

  // Workspace layout (f16 elements): activations A0..A5 then WT0..WT4.
  _Float16* ws = (_Float16*)d_ws;
  size_t off = 0;
  _Float16* Abuf[6];
  Abuf[0] = ws + off; off += (size_t)BATCH * KPAD[0];
  for (int i = 0; i < 5; ++i) { Abuf[i + 1] = ws + off; off += (size_t)BATCH * NPAD[i]; }
  _Float16* WT[5];
  for (int i = 0; i < 5; ++i) { WT[i] = ws + off; off += (size_t)NPAD[i] * KPAD[i]; }

  // 1) Weight prep (mask+transpose+pad to f16), every call (ws is volatile).
  for (int i = 0; i < 5; ++i) {
    size_t total = (size_t)NPAD[i] * KPAD[i];
    int blocks = (int)((total + 255) / 256);
    prep_weight_kernel<<<blocks, 256, 0, stream>>>(W[i], mk[i], WT[i],
                                                   FI[i], FO[i], KPAD[i], NPAD[i]);
  }

  // 2) Diagonal layer -> A0 (f16, zero-padded rows).
  {
    size_t total = (size_t)BATCH * KPAD[0];
    int blocks = (int)((total + 255) / 256);
    diag_kernel<<<blocks, 256, 0, stream>>>(x, w0, b0, Abuf[0], KPAD[0]);
  }

  // 3) Head 1 on A0.
  head_kernel<<<BATCH / 8, 256, 0, stream>>>(Abuf[0], KPAD[0], DIMS[0],
                                             lw[0], lb[0], out, 0);

  // 4) Masked GEMM chain with fused bias+tanh, each followed by its head.
  for (int i = 0; i < 5; ++i) {
    dim3 grid(NPAD[i] / TN, BATCH / TM);
    gemm_tanh_kernel<<<grid, 256, 0, stream>>>(Abuf[i], WT[i], bb[i], Abuf[i + 1],
                                               KPAD[i], NPAD[i], FO[i]);
    head_kernel<<<BATCH / 8, 256, 0, stream>>>(Abuf[i + 1], NPAD[i], DIMS[i + 1],
                                               lw[i + 1], lb[i + 1], out, i + 1);
  }
}
